// GAT_36215164240765
// MI455X (gfx1250) — compile-verified, matrix operand order checked by reading, not run on hardware
//
#include <hip/hip_runtime.h>
#include <hip/hip_bf16.h>

typedef __attribute__((ext_vector_type(16))) _Float16 v16h;
typedef __attribute__((ext_vector_type(8)))  _Float16 v8h;
typedef __attribute__((ext_vector_type(8)))  float    v8f;

// ---------------------------------------------------------------------------
// utility: zero fill
// ---------------------------------------------------------------------------
__global__ void gat_zero_u32(unsigned* __restrict__ p, long long n) {
    long long i = (long long)blockIdx.x * blockDim.x + threadIdx.x;
    if (i < n) p[i] = 0u;
}

// ---------------------------------------------------------------------------
// WMMA GEMM: C[M,N] = A[M,K] * B[K,N], f32 in/out, f16 multiply, f32 accum.
// Block = 128 threads (4 wave32s). Tile 64(M) x 64(N), K stepped by 32.
// Wave w owns rows [row0+16w, row0+16w+16) x all 64 cols (4 fragments).
//
// LDS layout exploits ISA 7.12.2: a 16x16 f16 tile held row-major means the
// WMMA fragment half for lane L is the 8 contiguous halfs at
// (row = L&15, col = (L>>4)*8)  ->  a single ds_load_b128 per K-half.
// A staged row-major [64][LDP]; B staged column-major [64][LDP] ([n][k]).
// LDP = 40 halfs (80 B): keeps 16B alignment, spreads rows across banks.
// ---------------------------------------------------------------------------
#define GT_TM 64
#define GT_TN 64
#define GT_TK 32
#define GT_LDP 40

union gat_frag { v16h v; v8h h[2]; };

__global__ __launch_bounds__(128)
void gat_wmma_gemm(const float* __restrict__ A, const float* __restrict__ B,
                   float* __restrict__ C, int M, int N, int K)
{
    __shared__ __align__(16) _Float16 As[GT_TM * GT_LDP];   // row-major  [m][k]
    __shared__ __align__(16) _Float16 Bs[GT_TN * GT_LDP];   // col-major  [n][k]

    const int tid  = threadIdx.x;
    const int lane = tid & 31;
    const int wave = tid >> 5;                 // 0..3
    const int row0 = blockIdx.x * GT_TM;
    const int col0 = blockIdx.y * GT_TN;

    const bool interior = (row0 + GT_TM <= M);   // uniform per block

    v8f acc[4] = {{}, {}, {}, {}};

    for (int k0 = 0; k0 < K; k0 += GT_TK) {
        // ---- stage A: 64 rows x 32 k = 256 chunks of 8 floats, 2 per thread
        if (interior) {
            #pragma unroll
            for (int it = 0; it < 2; ++it) {
                int chunk = tid + it * 128;
                int r  = chunk >> 2;
                int c0 = (chunk & 3) * 8;
                const float* gp = A + (size_t)(row0 + r) * K + (k0 + c0);
                v8h hv;
                #pragma unroll
                for (int j = 0; j < 8; ++j) hv[j] = (_Float16)gp[j];
                *(v8h*)&As[r * GT_LDP + c0] = hv;
            }
        } else {
            #pragma unroll
            for (int it = 0; it < 2; ++it) {
                int chunk = tid + it * 128;
                int r  = chunk >> 2;
                int c0 = (chunk & 3) * 8;
                v8h hv;
                if (row0 + r < M) {
                    const float* gp = A + (size_t)(row0 + r) * K + (k0 + c0);
                    #pragma unroll
                    for (int j = 0; j < 8; ++j) hv[j] = (_Float16)gp[j];
                } else {
                    #pragma unroll
                    for (int j = 0; j < 8; ++j) hv[j] = (_Float16)0.f;
                }
                *(v8h*)&As[r * GT_LDP + c0] = hv;
            }
        }
        // ---- stage B (transposed): 32 k x 64 n, 16 elems per thread
        #pragma unroll
        for (int it = 0; it < 16; ++it) {
            int i = tid + it * 128;
            int k = i >> 6, n = i & 63;
            int gk = k0 + k, gn = col0 + n;
            float v = (gk < K && gn < N) ? B[(size_t)gk * N + gn] : 0.0f;
            Bs[n * GT_LDP + k] = (_Float16)v;
        }
        __syncthreads();

        // ---- fragments: contiguous b128 LDS loads per ISA layout
        const int ml = lane & 15;
        const int hb = (lane >> 4) * 8;          // 0 or 8
        gat_frag af;
        const _Float16* ap = &As[(wave * 16 + ml) * GT_LDP];
        af.h[0] = *(const v8h*)(ap + hb);
        af.h[1] = *(const v8h*)(ap + 16 + hb);

        #pragma unroll
        for (int cf = 0; cf < 4; ++cf) {
            gat_frag bf;
            const _Float16* bp = &Bs[(cf * 16 + ml) * GT_LDP];
            bf.h[0] = *(const v8h*)(bp + hb);
            bf.h[1] = *(const v8h*)(bp + 16 + hb);
            acc[cf] = __builtin_amdgcn_wmma_f32_16x16x32_f16(
                          false, af.v, false, bf.v, (short)0, acc[cf], false, false);
        }
        __syncthreads();
    }

    // C/D layout: VGPR g -> lanes0-15: M=g, N=lane ; lanes16-31: M=8+g, N=lane-16
    const int nl   = lane & 15;
    const int mofs = (lane < 16) ? 0 : 8;
    #pragma unroll
    for (int cf = 0; cf < 4; ++cf) {
        int gn = col0 + cf * 16 + nl;
        if (gn < N) {
            #pragma unroll
            for (int g = 0; g < 8; ++g) {
                int gm = row0 + wave * 16 + mofs + g;
                if (gm < M) C[(size_t)gm * N + gn] = acc[cf][g];
            }
        }
    }
}

// ---------------------------------------------------------------------------
// alpha_src / alpha_dst: [N,H,C] x [H,C] -> [N,H]
// ---------------------------------------------------------------------------
__global__ void gat_alpha(const float* __restrict__ h,
                          const float* __restrict__ a_src,
                          const float* __restrict__ a_dst,
                          float* __restrict__ as, float* __restrict__ ad,
                          int n, int H, int C)
{
    int i = blockIdx.x * blockDim.x + threadIdx.x;   // node*H + head
    if (i >= n * H) return;
    int node = i / H, hh = i % H;
    const float* hp = h + (size_t)node * H * C + (size_t)hh * C;
    const float* sp = a_src + (size_t)hh * C;
    const float* dp = a_dst + (size_t)hh * C;
    float s = 0.f, d = 0.f;
    for (int c = 0; c < C; ++c) { float v = hp[c]; s += v * sp[c]; d += v * dp[c]; }
    as[i] = s; ad[i] = d;
}

// ---------------------------------------------------------------------------
// order-preserving float<->uint key for atomic max
// ---------------------------------------------------------------------------
__device__ __forceinline__ unsigned gat_fkey(float f) {
    unsigned b = __float_as_uint(f);
    return (b & 0x80000000u) ? ~b : (b | 0x80000000u);
}
__device__ __forceinline__ float gat_fkey_inv(unsigned k) {
    unsigned b = (k & 0x80000000u) ? (k & 0x7FFFFFFFu) : ~k;
    return __uint_as_float(b);
}
__device__ __forceinline__ void gat_edge_sd(const int* ei, int E, int e,
                                            int& src, int& dst) {
    if (e < E) { src = ei[e]; dst = ei[E + e]; }
    else       { src = e - E; dst = e - E; }       // self-loop
}

// pass 1: segment max of leaky_relu(as[src]+ad[dst])
__global__ void gat_edge_max(const int* __restrict__ ei, int E, int Nn,
                             const float* __restrict__ as,
                             const float* __restrict__ ad,
                             unsigned* __restrict__ mkey, int H)
{
    int e = blockIdx.x * blockDim.x + threadIdx.x;
    if (e >= E + Nn) return;
    int src, dst; gat_edge_sd(ei, E, e, src, dst);
    for (int hh = 0; hh < H; ++hh) {
        float v = as[src * H + hh] + ad[dst * H + hh];
        v = (v > 0.f) ? v : 0.2f * v;
        atomicMax(&mkey[dst * H + hh], gat_fkey(v));
    }
}

// pass 2: ex = exp(e - max[dst]); den[dst] += ex; stash ex per edge
__global__ void gat_edge_exp(const int* __restrict__ ei, int E, int Nn,
                             const float* __restrict__ as,
                             const float* __restrict__ ad,
                             const unsigned* __restrict__ mkey,
                             float* __restrict__ exbuf,
                             float* __restrict__ den, int H)
{
    int e = blockIdx.x * blockDim.x + threadIdx.x;
    if (e >= E + Nn) return;
    int src, dst; gat_edge_sd(ei, E, e, src, dst);
    for (int hh = 0; hh < H; ++hh) {
        float v = as[src * H + hh] + ad[dst * H + hh];
        v = (v > 0.f) ? v : 0.2f * v;
        float m  = gat_fkey_inv(mkey[dst * H + hh]);
        float ex = expf(v - m);
        exbuf[(size_t)e * H + hh] = ex;
        atomicAdd(&den[dst * H + hh], ex);
    }
}

// pass 3 (layer 1): 256 threads = one edge; feature f = h*32+c
__global__ __launch_bounds__(256)
void gat_edge_agg256(const int* __restrict__ ei, int E, int Nn,
                     const float* __restrict__ hfeat,
                     const float* __restrict__ exbuf,
                     const float* __restrict__ den,
                     float* __restrict__ agg)
{
    int e = blockIdx.x;
    if (e >= E + Nn) return;
    int src, dst; gat_edge_sd(ei, E, e, src, dst);
    int t  = threadIdx.x;            // 0..255
    int hh = t >> 5;                 // head
    float alpha = exbuf[(size_t)e * 8 + hh] / (den[dst * 8 + hh] + 1e-16f);
    float v = hfeat[(size_t)src * 256 + t] * alpha;
    atomicAdd(&agg[(size_t)dst * 256 + t], v);
}

// pass 3 (layer 2): 64 threads = one edge; 40 classes, 1 head
__global__ __launch_bounds__(64)
void gat_edge_agg40(const int* __restrict__ ei, int E, int Nn,
                    const float* __restrict__ hfeat,
                    const float* __restrict__ exbuf,
                    const float* __restrict__ den,
                    float* __restrict__ agg)
{
    int e = blockIdx.x;
    if (e >= E + Nn) return;
    int t = threadIdx.x;
    if (t >= 40) return;
    int src, dst; gat_edge_sd(ei, E, e, src, dst);
    float alpha = exbuf[e] / (den[dst] + 1e-16f);
    float v = hfeat[(size_t)src * 40 + t] * alpha;
    atomicAdd(&agg[(size_t)dst * 40 + t], v);
}

// h = elu(agg + b)  (elementwise)
__global__ void gat_elu_bias(const float* __restrict__ agg,
                             const float* __restrict__ b,
                             float* __restrict__ out, long long n, int F)
{
    long long i = (long long)blockIdx.x * blockDim.x + threadIdx.x;
    if (i >= n) return;
    float v = agg[i] + b[i % F];
    out[i] = (v > 0.f) ? v : (expf(v) - 1.0f);
}

// out = log_softmax(agg + b) over 40 classes per node
__global__ void gat_logsoftmax(const float* __restrict__ agg,
                               const float* __restrict__ b,
                               float* __restrict__ out, int n)
{
    int node = blockIdx.x * blockDim.x + threadIdx.x;
    if (node >= n) return;
    float vals[40];
    float m = -3.402823e38f;
    for (int c = 0; c < 40; ++c) {
        float v = agg[(size_t)node * 40 + c] + b[c];
        vals[c] = v; m = fmaxf(m, v);
    }
    float s = 0.f;
    for (int c = 0; c < 40; ++c) s += expf(vals[c] - m);
    float ls = m + logf(s);
    for (int c = 0; c < 40; ++c) out[(size_t)node * 40 + c] = vals[c] - ls;
}

// ---------------------------------------------------------------------------
extern "C" void kernel_launch(void* const* d_in, const int* in_sizes, int n_in,
                              void* d_out, int out_size, void* d_ws, size_t ws_size,
                              hipStream_t stream)
{
    const float* x      = (const float*)d_in[0];
    const int*   ei     = (const int*)  d_in[1];
    const float* W1     = (const float*)d_in[2];
    const float* a_src1 = (const float*)d_in[3];
    const float* a_dst1 = (const float*)d_in[4];
    const float* b1     = (const float*)d_in[5];
    const float* W2     = (const float*)d_in[6];
    const float* a_src2 = (const float*)d_in[7];
    const float* a_dst2 = (const float*)d_in[8];
    const float* b2     = (const float*)d_in[9];
    float* out = (float*)d_out;

    const int F_IN = 512, H = 8, C = 32, HC = 256, CLS = 40;
    const int N  = in_sizes[0] / F_IN;     // 50000
    const int E  = in_sizes[1] / 2;        // 800000
    const int ET = E + N;                  // + self-loops

    // workspace carve-up (all f32-sized slots; ~154 MB total)
    float* ws = (float*)d_ws;
    size_t o = 0;
    float*    h1    = ws + o; o += (size_t)N * HC;
    float*    as1   = ws + o; o += (size_t)N * H;
    float*    ad1   = ws + o; o += (size_t)N * H;
    unsigned* mkey1 = (unsigned*)(ws + o); o += (size_t)N * H;
    float*    den1  = ws + o; o += (size_t)N * H;
    float*    ex1   = ws + o; o += (size_t)ET * H;
    float*    agg1  = ws + o; o += (size_t)N * HC;
    float*    y2    = ws + o; o += (size_t)N * CLS;
    float*    as2   = ws + o; o += (size_t)N;
    float*    ad2   = ws + o; o += (size_t)N;
    unsigned* mkey2 = (unsigned*)(ws + o); o += (size_t)N;
    float*    den2  = ws + o; o += (size_t)N;
    float*    ex2   = ws + o; o += (size_t)ET;
    float*    agg2  = ws + o; o += (size_t)N * CLS;

    auto zero = [&](void* p, long long cnt) {
        gat_zero_u32<<<(int)((cnt + 511) / 512), 512, 0, stream>>>((unsigned*)p, cnt);
    };
    zero(mkey1, (long long)N * H);
    zero(den1,  (long long)N * H);
    zero(agg1,  (long long)N * HC);
    zero(mkey2, N);
    zero(den2,  N);
    zero(agg2,  (long long)N * CLS);

    // ---- layer 1 ----
    {
        dim3 grid((N + GT_TM - 1) / GT_TM, HC / GT_TN);
        gat_wmma_gemm<<<grid, 128, 0, stream>>>(x, W1, h1, N, HC, F_IN);
    }
    gat_alpha<<<(N * H + 255) / 256, 256, 0, stream>>>(h1, a_src1, a_dst1, as1, ad1, N, H, C);
    gat_edge_max<<<(ET + 255) / 256, 256, 0, stream>>>(ei, E, N, as1, ad1, mkey1, H);
    gat_edge_exp<<<(ET + 255) / 256, 256, 0, stream>>>(ei, E, N, as1, ad1, mkey1, ex1, den1, H);
    gat_edge_agg256<<<ET, 256, 0, stream>>>(ei, E, N, h1, ex1, den1, agg1);
    gat_elu_bias<<<(int)(((long long)N * HC + 255) / 256), 256, 0, stream>>>(
        agg1, b1, agg1, (long long)N * HC, HC);

    // ---- layer 2 ----
    {
        dim3 grid((N + GT_TM - 1) / GT_TM, (CLS + GT_TN - 1) / GT_TN);
        gat_wmma_gemm<<<grid, 128, 0, stream>>>(agg1, W2, y2, N, CLS, HC);
    }
    gat_alpha<<<(N + 255) / 256, 256, 0, stream>>>(y2, a_src2, a_dst2, as2, ad2, N, 1, CLS);
    gat_edge_max<<<(ET + 255) / 256, 256, 0, stream>>>(ei, E, N, as2, ad2, mkey2, 1);
    gat_edge_exp<<<(ET + 255) / 256, 256, 0, stream>>>(ei, E, N, as2, ad2, mkey2, ex2, den2, 1);
    gat_edge_agg40<<<ET, 64, 0, stream>>>(ei, E, N, y2, ex2, den2, agg2);

    gat_logsoftmax<<<(N + 127) / 128, 128, 0, stream>>>(agg2, b2, out, N);
}
